// TransformerConvLayer_55903294325151
// MI455X (gfx1250) — matrix operand order repaired and checked by reading, not compiled, not run on hardware
//
#include <hip/hip_runtime.h>
#include <hip/hip_bf16.h>

#define NN 50000
#define NE 400000
#define DM 128
#define NH 4
#define HD (DM * NH)   // 512
#define ETOT (NE + NN) // 450000

typedef __attribute__((ext_vector_type(16))) __bf16 v16bf;
typedef __attribute__((ext_vector_type(8)))  __bf16 v8bf;
typedef __attribute__((ext_vector_type(8)))  float  v8f;

static __device__ inline __bf16 f32_to_bf16(float f) {
    unsigned u = __float_as_uint(f);
    unsigned r = u + 0x7FFFu + ((u >> 16) & 1u);   // RNE
    unsigned short h = (unsigned short)(r >> 16);
    __bf16 out;
    __builtin_memcpy(&out, &h, 2);
    return out;
}
static __device__ inline float bf16_to_f32(__bf16 b) {
    unsigned short h;
    __builtin_memcpy(&h, &b, 2);
    return __uint_as_float(((unsigned)h) << 16);
}
static __device__ inline v16bf cat8(v8bf a, v8bf b) {
    return __builtin_shufflevector(a, b, 0, 1, 2, 3, 4, 5, 6, 7,
                                         8, 9, 10, 11, 12, 13, 14, 15);
}

// ---------------- prep: x[N,128] f32 -> xhi/xlo bf16 (row-major) ----------------
__global__ void __launch_bounds__(256)
cvt_x_kernel(const float* __restrict__ x, __bf16* __restrict__ xhi,
             __bf16* __restrict__ xlo) {
    const int i = blockIdx.x * blockDim.x + threadIdx.x;
    if (i >= NN * DM) return;
    float v = x[i];
    __bf16 hi = f32_to_bf16(v);
    xhi[i] = hi;
    xlo[i] = f32_to_bf16(v - bf16_to_f32(hi));
}

// ---------------- prep: W[128,512] f32 -> wt_hi/wt_lo bf16 transposed [512,128] ----------------
__global__ void __launch_bounds__(256)
cvt_w_kernel(const float* __restrict__ W, __bf16* __restrict__ wthi,
             __bf16* __restrict__ wtlo) {
    const int i = blockIdx.x * blockDim.x + threadIdx.x;
    if (i >= HD * DM) return;
    const int col = i / DM;     // 0..511
    const int k   = i - col * DM;
    float v = W[(size_t)k * HD + col];
    __bf16 hi = f32_to_bf16(v);
    wthi[(size_t)col * DM + k] = hi;
    wtlo[(size_t)col * DM + k] = f32_to_bf16(v - bf16_to_f32(hi));
}

// ---------------- GEMM: h[N,512] = x @ W, bf16x3 WMMA, pre-converted operands ----------------
__global__ void __launch_bounds__(32)
gat_gemm_wmma(const __bf16* __restrict__ xhi, const __bf16* __restrict__ xlo,
              const __bf16* __restrict__ wthi, const __bf16* __restrict__ wtlo,
              float* __restrict__ h) {
    const int row0 = blockIdx.x * 16;
    const int col0 = blockIdx.y * 128;
    const int lane = threadIdx.x;
    const int half = lane >> 4;
    const int rl   = lane & 15;

    // A fragments: element e -> k = ks*32 + (e>>3)*16 + half*8 + (e&7)
    // -> two contiguous 8-bf16 (16B) loads per fragment
    v16bf a_hi[4], a_lo[4];
    const __bf16* xh = xhi + (size_t)(row0 + rl) * DM;
    const __bf16* xl = xlo + (size_t)(row0 + rl) * DM;
#pragma unroll
    for (int ks = 0; ks < 4; ++ks) {
        const int o0 = ks * 32 + half * 8;
        const int o1 = o0 + 16;
        a_hi[ks] = cat8(*(const v8bf*)(xh + o0), *(const v8bf*)(xh + o1));
        a_lo[ks] = cat8(*(const v8bf*)(xl + o0), *(const v8bf*)(xl + o1));
    }

#pragma unroll
    for (int nt = 0; nt < 8; ++nt) {
        const int col = col0 + nt * 16 + rl;
        const __bf16* wh = wthi + (size_t)col * DM;
        const __bf16* wl = wtlo + (size_t)col * DM;
        v8f c = {};
#pragma unroll
        for (int ks = 0; ks < 4; ++ks) {
            // B fragment: element e -> k = ks*32 + (e>>3)*16 + half*8 + (e&7)
            const int ko = ks * 32 + half * 8;
            v16bf bh = cat8(*(const v8bf*)(wh + ko), *(const v8bf*)(wh + ko + 16));
            v16bf bl = cat8(*(const v8bf*)(wl + ko), *(const v8bf*)(wl + ko + 16));
            c = __builtin_amdgcn_wmma_f32_16x16x32_bf16(false, a_hi[ks], false, bh, (short)0, c, false, false);
            c = __builtin_amdgcn_wmma_f32_16x16x32_bf16(false, a_lo[ks], false, bh, (short)0, c, false, false);
            c = __builtin_amdgcn_wmma_f32_16x16x32_bf16(false, a_hi[ks], false, bl, (short)0, c, false, false);
        }
#pragma unroll
        for (int r = 0; r < 8; ++r) {
            h[(size_t)(row0 + (half << 3) + r) * HD + col] = c[r];
        }
    }
}

// ---------------- a_src/a_dst: [N,H] per-head dot products ----------------
__global__ void __launch_bounds__(128)
gat_attn_proj(const float* __restrict__ h, const float* __restrict__ att_src,
              const float* __restrict__ att_dst, float* __restrict__ a_src,
              float* __restrict__ a_dst) {
    const int n = blockIdx.x;
    const int head = threadIdx.x >> 5;
    const int lane = threadIdx.x & 31;
    const float* hp = h + (size_t)n * HD + head * DM + lane * 4;
    const float* sp = att_src + head * DM + lane * 4;
    const float* dp = att_dst + head * DM + lane * 4;
    float s = 0.f, d = 0.f;
#pragma unroll
    for (int j = 0; j < 4; ++j) { s += hp[j] * sp[j]; d += hp[j] * dp[j]; }
#pragma unroll
    for (int off = 16; off > 0; off >>= 1) {
        s += __shfl_down(s, off, 32);
        d += __shfl_down(d, off, 32);
    }
    if (lane == 0) {
        a_src[n * NH + head] = s;
        a_dst[n * NH + head] = d;
    }
}

// ---------------- init workspace (m=-inf, z=0, accum=0) ----------------
__global__ void gat_init(float* __restrict__ m, float* __restrict__ z,
                         float* __restrict__ accum) {
    const int stride = gridDim.x * blockDim.x;
    const float ninf = __uint_as_float(0xFF800000u);
    for (int i = blockIdx.x * blockDim.x + threadIdx.x; i < NN * DM; i += stride)
        accum[i] = 0.f;
    for (int i = blockIdx.x * blockDim.x + threadIdx.x; i < NN * NH; i += stride) {
        m[i] = ninf;
        z[i] = 0.f;
    }
}

static __device__ inline void atomicMaxF32(float* addr, float val) {
    if (val >= 0.f) atomicMax((int*)addr, __float_as_int(val));
    else            atomicMin((unsigned int*)addr, (unsigned int)__float_as_int(val));
}

// ---------------- edge pass 1: e = leakyrelu(as[src]+ad[dst]); segment max ----------------
__global__ void __launch_bounds__(256)
gat_edge_max(const int* __restrict__ ei, const float* __restrict__ a_src,
             const float* __restrict__ a_dst, float* __restrict__ ew,
             float* __restrict__ m) {
    const int i = blockIdx.x * blockDim.x + threadIdx.x;
    if (i >= ETOT) return;
    int s, d;
    if (i < NE) { s = ei[i]; d = ei[NE + i]; } else { s = d = i - NE; }
    if ((unsigned)s >= (unsigned)NN) s = 0;
    if ((unsigned)d >= (unsigned)NN) d = 0;
#pragma unroll
    for (int hd = 0; hd < NH; ++hd) {
        float e = a_src[s * NH + hd] + a_dst[d * NH + hd];
        e = (e > 0.f) ? e : 0.2f * e;
        ew[(size_t)i * NH + hd] = e;
        atomicMaxF32(&m[d * NH + hd], e);
    }
}

// ---------------- edge pass 2: w = exp(e - m[dst]); segment sum ----------------
__global__ void __launch_bounds__(256)
gat_edge_sum(const int* __restrict__ ei, const float* __restrict__ m,
             float* __restrict__ ew, float* __restrict__ z) {
    const int i = blockIdx.x * blockDim.x + threadIdx.x;
    if (i >= ETOT) return;
    int d = (i < NE) ? ei[NE + i] : (i - NE);
    if ((unsigned)d >= (unsigned)NN) d = 0;
#pragma unroll
    for (int hd = 0; hd < NH; ++hd) {
        float w = __expf(ew[(size_t)i * NH + hd] - m[d * NH + hd]);
        ew[(size_t)i * NH + hd] = w;
        atomicAdd(&z[d * NH + hd], w);
    }
}

// ---------------- edge pass 3: accum[dst,c] += sum_h alpha_h * h[src,h,c] ----------------
__global__ void __launch_bounds__(256)
gat_edge_aggr(const int* __restrict__ ei, const float* __restrict__ h,
              const float* __restrict__ ew, const float* __restrict__ z,
              float* __restrict__ accum) {
    const int wid  = (blockIdx.x * blockDim.x + threadIdx.x) >> 5; // edge id
    const int lane = threadIdx.x & 31;
    if (wid >= ETOT) return;
    int s, d;
    if (wid < NE) { s = ei[wid]; d = ei[NE + wid]; } else { s = d = wid - NE; }
    if ((unsigned)s >= (unsigned)NN) s = 0;
    if ((unsigned)d >= (unsigned)NN) d = 0;

    float alpha[NH];
#pragma unroll
    for (int hd = 0; hd < NH; ++hd)
        alpha[hd] = ew[(size_t)wid * NH + hd] / z[d * NH + hd];

    const int c = lane * 4;
    float acc0 = 0.f, acc1 = 0.f, acc2 = 0.f, acc3 = 0.f;
#pragma unroll
    for (int hd = 0; hd < NH; ++hd) {
        const float* hp = h + (size_t)s * HD + hd * DM + c;
        acc0 += alpha[hd] * hp[0];
        acc1 += alpha[hd] * hp[1];
        acc2 += alpha[hd] * hp[2];
        acc3 += alpha[hd] * hp[3];
    }
    float* ap = accum + (size_t)d * DM + c;
    atomicAdd(ap + 0, acc0);
    atomicAdd(ap + 1, acc1);
    atomicAdd(ap + 2, acc2);
    atomicAdd(ap + 3, acc3);
}

// ---------------- finalize: /H + bias -> LayerNorm -> ReLU -> +x ----------------
__global__ void __launch_bounds__(256)
gat_finalize(const float* __restrict__ accum, const float* __restrict__ bias,
             const float* __restrict__ gamma, const float* __restrict__ beta,
             const float* __restrict__ x, float* __restrict__ out) {
    const int n    = (blockIdx.x * blockDim.x + threadIdx.x) >> 5; // node
    const int lane = threadIdx.x & 31;
    if (n >= NN) return;
    const int c = lane * 4;
    float v[4];
    float s = 0.f, s2 = 0.f;
#pragma unroll
    for (int j = 0; j < 4; ++j) {
        float t = accum[(size_t)n * DM + c + j] * 0.25f + bias[c + j];
        v[j] = t;
        s += t;
        s2 += t * t;
    }
#pragma unroll
    for (int off = 16; off > 0; off >>= 1) {
        s  += __shfl_xor(s, off, 32);
        s2 += __shfl_xor(s2, off, 32);
    }
    const float mu  = s * (1.0f / DM);
    const float var = s2 * (1.0f / DM) - mu * mu;
    const float inv = rsqrtf(var + 1e-5f);
#pragma unroll
    for (int j = 0; j < 4; ++j) {
        float ln = gamma[c + j] * (v[j] - mu) * inv + beta[c + j];
        float r  = (ln > 0.f ? ln : 0.f) + x[(size_t)n * DM + c + j];
        out[(size_t)n * DM + c + j] = r;
    }
}

extern "C" void kernel_launch(void* const* d_in, const int* in_sizes, int n_in,
                              void* d_out, int out_size, void* d_ws, size_t ws_size,
                              hipStream_t stream) {
    const float* x        = (const float*)d_in[0];
    const int*   ei       = (const int*)d_in[1];
    const float* W        = (const float*)d_in[2];
    const float* att_src  = (const float*)d_in[3];
    const float* att_dst  = (const float*)d_in[4];
    const float* bias     = (const float*)d_in[5];
    const float* ln_gamma = (const float*)d_in[6];
    const float* ln_beta  = (const float*)d_in[7];
    float*       out      = (float*)d_out;

    // workspace carve-up (floats)
    float* ws    = (float*)d_ws;
    float* h     = ws;                                   // N*512  = 25,600,000
    float* asrc  = h    + (size_t)NN * HD;               // N*4
    float* adst  = asrc + (size_t)NN * NH;               // N*4
    float* mbuf  = adst + (size_t)NN * NH;               // N*4
    float* zbuf  = mbuf + (size_t)NN * NH;               // N*4
    float* ew    = zbuf + (size_t)NN * NH;               // (E+N)*4 = 1,800,000
    float* accum = ew   + (size_t)ETOT * NH;             // N*128  = 6,400,000

    // bf16 staging buffers ALIAS the ew+accum region (8.2M floats = 32.8MB):
    // only live during prep+gemm, which run before init/edge passes.
    __bf16* xhi  = (__bf16*)ew;                          // N*128 bf16 = 12.8MB
    __bf16* xlo  = xhi + (size_t)NN * DM;                // N*128 bf16
    __bf16* wthi = xlo + (size_t)NN * DM;                // 512*128 bf16
    __bf16* wtlo = wthi + (size_t)HD * DM;               // 512*128 bf16
    (void)in_sizes; (void)n_in; (void)out_size; (void)ws_size;

    // 1) convert operands to bf16 hi/lo (split precision)
    cvt_x_kernel<<<(NN * DM + 255) / 256, 256, 0, stream>>>(x, xhi, xlo);
    cvt_w_kernel<<<(HD * DM + 255) / 256, 256, 0, stream>>>(W, wthi, wtlo);

    // 2) feature projection via bf16x3 WMMA (pure loads + wmma)
    gat_gemm_wmma<<<dim3(NN / 16, HD / 128), 32, 0, stream>>>(xhi, xlo, wthi, wtlo, h);

    // 3) per-node attention coefficients
    gat_attn_proj<<<NN, 128, 0, stream>>>(h, att_src, att_dst, asrc, adst);

    // 4) init m/z/accum (after gemm: accum region aliased the bf16 staging)
    gat_init<<<4096, 256, 0, stream>>>(mbuf, zbuf, accum);

    // 5) segment softmax over incoming edges
    const int eb = (ETOT + 255) / 256;
    gat_edge_max<<<eb, 256, 0, stream>>>(ei, asrc, adst, ew, mbuf);
    gat_edge_sum<<<eb, 256, 0, stream>>>(ei, mbuf, ew, zbuf);

    // 6) weighted aggregation (one wave per edge, head-mean folded in)
    gat_edge_aggr<<<(ETOT + 7) / 8, 256, 0, stream>>>(ei, h, ew, zbuf, accum);

    // 7) bias + LayerNorm + ReLU + residual
    gat_finalize<<<(NN + 7) / 8, 256, 0, stream>>>(accum, bias, ln_gamma, ln_beta, x, out);
}
